// PAM_Maps_Diff_8057358647936
// MI455X (gfx1250) — matrix operand order repaired and checked by reading, not compiled, not run on hardware
//
#include <hip/hip_runtime.h>
#include <hip/hip_bf16.h>
#include <stdint.h>

typedef __attribute__((ext_vector_type(16))) __bf16 v16bf;
typedef __attribute__((ext_vector_type(8)))  float  v8f;
typedef __attribute__((ext_vector_type(4)))  int    v4i;

static constexpr int kB   = 2;
static constexpr int kCm  = 6;
static constexpr int kC   = 64;
static constexpr int kH   = 96;
static constexpr int kW   = 96;
static constexpr int kHW  = kH * kW;          // 9216
static constexpr int kNT  = kHW / 16;         // 576 n-tiles per batch

static constexpr int kMStage  = 128;          // m-values staged in LDS per round
static constexpr int kStages  = kHW / kMStage;// 72
static constexpr int kFdPitch = 68;           // dwords per staged fd row (64 + 4 pad)

__device__ __forceinline__ uint32_t pack2_bf16(float a, float b) {
  uint32_t ua = __float_as_uint(a), ub = __float_as_uint(b);
  uint32_t ha = (ua + 0x7FFFu + ((ua >> 16) & 1u)) >> 16;
  uint32_t hb = (ub + 0x7FFFu + ((ub >> 16) & 1u)) >> 16;
  return (ha & 0xFFFFu) | (hb << 16);
}

// xor-16 half-wave swap: v_permlanex16_b32 (VALU, no LDS round trip)
__device__ __forceinline__ uint32_t xor16_u32(uint32_t v) {
#if __has_builtin(__builtin_amdgcn_permlanex16)
  return __builtin_amdgcn_permlanex16(v, v, 0x76543210u, 0xfedcba98u, false, false);
#else
  return __shfl_xor(v, 16, 32);
#endif
}
__device__ __forceinline__ float xor16_f32(float v) {
  return __uint_as_float(xor16_u32(__float_as_uint(v)));
}

// 16B global -> LDS async copy (ASYNCcnt-tracked); register-copy fallback.
// Builtin signature (from diagnostic): (AS1 v4i* gsrc, AS3 v4i* ldst, imm, imm).
__device__ __forceinline__ void async_copy16(uint32_t* lds_dst, const uint32_t* gsrc) {
#if __has_builtin(__builtin_amdgcn_global_load_async_to_lds_b128)
  __builtin_amdgcn_global_load_async_to_lds_b128(
      (__attribute__((address_space(1))) v4i*)gsrc,
      (__attribute__((address_space(3))) v4i*)lds_dst, 0, 0);
#else
  *(uint4*)lds_dst = *(const uint4*)gsrc;
#endif
}
__device__ __forceinline__ void wait_async0() {
#if __has_builtin(__builtin_amdgcn_global_load_async_to_lds_b128)
  asm volatile("s_wait_asynccnt 0x0" ::: "memory");
#endif
}

__device__ __forceinline__ float wave_max32(float v) {
  #pragma unroll
  for (int m = 16; m >= 1; m >>= 1) v = fmaxf(v, __shfl_xor(v, m, 32));
  return v;
}
__device__ __forceinline__ float wave_sum32(float v) {
  #pragma unroll
  for (int m = 16; m >= 1; m >>= 1) v += __shfl_xor(v, m, 32);
  return v;
}

// ---------------------------------------------------------------------------
// Kernel 1: diff = |softmax(softmax(map1) - softmax(map2))| over W.
// One wave per (b, c, h) row of 96 elements; 3 elements per lane.
// ---------------------------------------------------------------------------
__global__ __launch_bounds__(256)
void pam_diff_kernel(const float* __restrict__ m1, const float* __restrict__ m2,
                     float* __restrict__ diff) {
  const int wid  = blockIdx.x * 8 + (threadIdx.x >> 5);
  const int lane = threadIdx.x & 31;
  if (wid >= kB * kCm * kH) return;
  const float* r1 = m1 + (size_t)wid * kW;
  const float* r2 = m2 + (size_t)wid * kW;

  float x1[3], x2[3];
  #pragma unroll
  for (int j = 0; j < 3; ++j) {
    x1[j] = r1[lane + 32 * j];
    x2[j] = r2[lane + 32 * j];
  }
  float mx1 = wave_max32(fmaxf(fmaxf(x1[0], x1[1]), x1[2]));
  float mx2 = wave_max32(fmaxf(fmaxf(x2[0], x2[1]), x2[2]));
  float s1 = 0.f, s2 = 0.f;
  #pragma unroll
  for (int j = 0; j < 3; ++j) {
    x1[j] = __expf(x1[j] - mx1); s1 += x1[j];
    x2[j] = __expf(x2[j] - mx2); s2 += x2[j];
  }
  s1 = wave_sum32(s1); s2 = wave_sum32(s2);
  const float i1 = 1.f / s1, i2 = 1.f / s2;

  float d[3];
  #pragma unroll
  for (int j = 0; j < 3; ++j) d[j] = x1[j] * i1 - x2[j] * i2;
  float mxd = wave_max32(fmaxf(fmaxf(d[0], d[1]), d[2]));
  float sd = 0.f;
  #pragma unroll
  for (int j = 0; j < 3; ++j) { d[j] = __expf(d[j] - mxd); sd += d[j]; }
  sd = wave_sum32(sd);
  const float id = 1.f / sd;
  float* ro = diff + (size_t)wid * kW;
  #pragma unroll
  for (int j = 0; j < 3; ++j) ro[lane + 32 * j] = fabsf(d[j] * id);
}

// ---------------------------------------------------------------------------
// Kernel 2a: feat_b/feat_c as zero-padded bf16 rows [B, HW, 8] (16B/pixel).
// ---------------------------------------------------------------------------
__global__ __launch_bounds__(256)
void pam_proj_bc_kernel(const float* __restrict__ diff,
                        const float* __restrict__ Wb, const float* __restrict__ bb,
                        const float* __restrict__ Wc, const float* __restrict__ bc,
                        uint4* __restrict__ fbT8, uint4* __restrict__ fcT8) {
  const int t = blockIdx.x * blockDim.x + threadIdx.x;
  if (t >= kB * kHW) return;
  const int b = t / kHW, p = t % kHW;

  float d[kCm];
  #pragma unroll
  for (int k = 0; k < kCm; ++k) d[k] = diff[((size_t)b * kCm + k) * kHW + p];

  float ob[kCm], oc[kCm];
  #pragma unroll
  for (int o = 0; o < kCm; ++o) {
    float ab = bb[o], ac = bc[o];
    #pragma unroll
    for (int k = 0; k < kCm; ++k) {
      ab += Wb[o * kCm + k] * d[k];
      ac += Wc[o * kCm + k] * d[k];
    }
    ob[o] = ab; oc[o] = ac;
  }
  uint4 vb, vc;
  vb.x = pack2_bf16(ob[0], ob[1]); vb.y = pack2_bf16(ob[2], ob[3]);
  vb.z = pack2_bf16(ob[4], ob[5]); vb.w = 0u;
  vc.x = pack2_bf16(oc[0], oc[1]); vc.y = pack2_bf16(oc[2], oc[3]);
  vc.z = pack2_bf16(oc[4], oc[5]); vc.w = 0u;
  fbT8[t] = vb; fcT8[t] = vc;
}

// ---------------------------------------------------------------------------
// Kernel 2b: feat_d = Wd @ feature_map + bd, bf16 [B, 64, HW].
// ---------------------------------------------------------------------------
__global__ __launch_bounds__(256)
void pam_proj_d_kernel(const float* __restrict__ fm, const float* __restrict__ Wd,
                       const float* __restrict__ bd, uint16_t* __restrict__ fd) {
  const int t = blockIdx.x * blockDim.x + threadIdx.x;
  if (t >= kB * kC * kHW) return;
  const int p = t % kHW;
  const int o = (t / kHW) % kC;
  const int b = t / (kHW * kC);

  float acc = bd[o];
  const float* fmb = fm + (size_t)b * kC * kHW + p;
  #pragma unroll 8
  for (int c = 0; c < kC; ++c) acc += Wd[o * kC + c] * fmb[(size_t)c * kHW];

  uint32_t u = __float_as_uint(acc);
  fd[t] = (uint16_t)((u + 0x7FFFu + ((u >> 16) & 1u)) >> 16);
}

// ---------------------------------------------------------------------------
// Kernel 3: fused flash attention with double-buffered async LDS staging.
// ---------------------------------------------------------------------------
__global__ __launch_bounds__(256)
void pam_attn_kernel(const uint4* __restrict__ fbT8, const uint4* __restrict__ fcT8,
                     const uint32_t* __restrict__ fd, const float* __restrict__ fmap,
                     const float* __restrict__ alpha, float* __restrict__ out) {
  __shared__ __align__(16) uint32_t s_fd[2][kC * kFdPitch];  // 2 x 17408 B
  __shared__ __align__(16) uint4    s_fc[2][kMStage];        // 2 x 2048 B

  const int tid  = threadIdx.x;
  const int lane = tid & 31;
  const int wid  = blockIdx.x * 8 + (tid >> 5);
  const int b    = wid / kNT;
  const int n0   = (wid % kNT) * 16;
  const int half = lane >> 4;
  const int l16  = lane & 15;

  union U16 { v16bf bf; uint32_t u[8]; uint4 q[2]; };
  union U8  { v8f v; float x[8]; };

  const uint4*    fcB = fcT8 + (size_t)b * kHW;
  const uint32_t* fdB = fd + (size_t)b * kC * (kHW / 2);

  // stage loader: 1024 fd blocks + 128 fc blocks of 16B across 256 threads
  auto issue_stage = [&](int s, int buf) {
    const int m0s = s * kMStage;
    #pragma unroll
    for (int i = 0; i < 4; ++i) {
      const int q   = tid + 256 * i;
      const int row = q >> 4;          // 0..63
      const int j   = q & 15;          // 16B block within row
      async_copy16(&s_fd[buf][row * kFdPitch + j * 4],
                   fdB + (size_t)row * (kHW / 2) + (m0s >> 1) + j * 4);
    }
    if (tid < kMStage)
      async_copy16((uint32_t*)&s_fc[buf][tid],
                   (const uint32_t*)(fcB + m0s + tid));
  };

  // B-layout Q tile: lanes 0-15 hold channel pairs K=0..7 (c6,c7 zero-padded);
  // all other K positions stay zero.
  U16 bQ;
  #pragma unroll
  for (int i = 0; i < 8; ++i) bQ.u[i] = 0u;
  if (half == 0) bQ.q[0] = fbT8[(size_t)b * kHW + n0 + l16];

  U8 o[4];
  #pragma unroll
  for (int cb = 0; cb < 4; ++cb)
    #pragma unroll
    for (int r = 0; r < 8; ++r) o[cb].x[r] = 0.f;

  // A-layout fc tiles: zero once; only q[0] of lanes 0-15 is refreshed.
  U16 aA, aB;
  #pragma unroll
  for (int i = 0; i < 8; ++i) { aA.u[i] = 0u; aB.u[i] = 0u; }

  float mrun = -INFINITY, lrun = 0.f;
  const v8f zc = {};

  issue_stage(0, 0);
  wait_async0();
  __syncthreads();

  #pragma unroll 1
  for (int s = 0; s < kStages; ++s) {
    const int buf = s & 1;
    if (s + 1 < kStages) issue_stage(s + 1, buf ^ 1);

    #pragma unroll 1
    for (int ch = 0; ch < kMStage / 32; ++ch) {
      const int m0l = ch * 32;
      if (half == 0) {
        aA.q[0] = s_fc[buf][m0l + l16];
        aB.q[0] = s_fc[buf][m0l + 16 + l16];
      }

      U8 sA, sB;
      sA.v = __builtin_amdgcn_wmma_f32_16x16x32_bf16(false, aA.bf, false, bQ.bf,
                                                     (short)0, zc, false, false);
      sB.v = __builtin_amdgcn_wmma_f32_16x16x32_bf16(false, aB.bf, false, bQ.bf,
                                                     (short)0, zc, false, false);

      // ---- online softmax over this chunk's 32 m-rows (per column n)
      float tmax = sA.x[0];
      #pragma unroll
      for (int r = 1; r < 8; ++r) tmax = fmaxf(tmax, sA.x[r]);
      #pragma unroll
      for (int r = 0; r < 8; ++r) tmax = fmaxf(tmax, sB.x[r]);
      tmax = fmaxf(tmax, xor16_f32(tmax));

      const float mnew = fmaxf(mrun, tmax);
      if (__any(tmax > mrun)) {          // skip rescale once max settles
        const float scale = __expf(mrun - mnew);
        lrun *= scale;
        #pragma unroll
        for (int cb = 0; cb < 4; ++cb)
          #pragma unroll
          for (int r = 0; r < 8; ++r) o[cb].x[r] *= scale;
      }
      mrun = mnew;

      float psum = 0.f;
      #pragma unroll
      for (int r = 0; r < 8; ++r) { sA.x[r] = __expf(sA.x[r] - mnew); psum += sA.x[r]; }
      #pragma unroll
      for (int r = 0; r < 8; ++r) { sB.x[r] = __expf(sB.x[r] - mnew); psum += sB.x[r]; }
      psum += xor16_f32(psum);
      lrun += psum;

      // ---- convert Pᵀ (C-layout) into B-layout (32m x 16n) bf16
      uint32_t pkA[4], pkB[4];
      #pragma unroll
      for (int j = 0; j < 4; ++j) {
        pkA[j] = pack2_bf16(sA.x[2 * j], sA.x[2 * j + 1]);
        pkB[j] = pack2_bf16(sB.x[2 * j], sB.x[2 * j + 1]);
      }
      U16 bP;
      #pragma unroll
      for (int j = 0; j < 4; ++j) {
        const uint32_t swA = xor16_u32(pkA[j]);
        const uint32_t swB = xor16_u32(pkB[j]);
        bP.u[j]     = (half == 0) ? pkA[j] : swB;
        bP.u[4 + j] = (half == 0) ? swA : pkB[j];
      }

      // ---- Oᵀ += fd-tile @ Pᵀ, 4 channel blocks of 16 (A-tiles from LDS)
      #pragma unroll
      for (int cb = 0; cb < 4; ++cb) {
        const uint32_t* prow =
            &s_fd[buf][(cb * 16 + l16) * kFdPitch + (m0l >> 1) + 8 * half];
        U16 aD;
        aD.q[0] = *(const uint4*)(prow);
        aD.q[1] = *(const uint4*)(prow + 4);
        o[cb].v = __builtin_amdgcn_wmma_f32_16x16x32_bf16(false, aD.bf, false, bP.bf,
                                                          (short)0, o[cb].v, false, false);
      }
    }

    wait_async0();
    __syncthreads();
  }

  // ---- epilogue: normalize, alpha-scale, residual add
  const float linv = 1.f / lrun;
  const float al = alpha[0];
  #pragma unroll
  for (int cb = 0; cb < 4; ++cb) {
    #pragma unroll
    for (int r = 0; r < 8; ++r) {
      const int c = cb * 16 + r + 8 * half;
      const size_t idx = ((size_t)b * kC + c) * kHW + n0 + l16;
      out[idx] = al * (o[cb].x[r] * linv) + fmap[idx];
    }
  }
}

// ---------------------------------------------------------------------------
extern "C" void kernel_launch(void* const* d_in, const int* in_sizes, int n_in,
                              void* d_out, int out_size, void* d_ws, size_t ws_size,
                              hipStream_t stream) {
  const float* map1  = (const float*)d_in[0];
  const float* map2  = (const float*)d_in[1];
  const float* fmap  = (const float*)d_in[2];
  const float* Wb    = (const float*)d_in[3];
  const float* bb    = (const float*)d_in[4];
  const float* Wc    = (const float*)d_in[5];
  const float* bc    = (const float*)d_in[6];
  const float* Wd    = (const float*)d_in[7];
  const float* bd    = (const float*)d_in[8];
  const float* alpha = (const float*)d_in[9];
  float* out = (float*)d_out;

  char* ws = (char*)d_ws;
  const size_t diff_bytes = (size_t)kB * kCm * kHW * sizeof(float);   // 442368
  const size_t bc8_bytes  = (size_t)kB * kHW * 16;                    // 294912
  float*    diff = (float*)ws;
  uint4*    fbT8 = (uint4*)(ws + diff_bytes);
  uint4*    fcT8 = (uint4*)(ws + diff_bytes + bc8_bytes);
  uint16_t* fd   = (uint16_t*)(ws + diff_bytes + 2 * bc8_bytes);

  pam_diff_kernel<<<(kB * kCm * kH + 7) / 8, 256, 0, stream>>>(map1, map2, diff);
  pam_proj_bc_kernel<<<(kB * kHW + 255) / 256, 256, 0, stream>>>(diff, Wb, bb, Wc, bc,
                                                                 fbT8, fcT8);
  pam_proj_d_kernel<<<(kB * kC * kHW + 255) / 256, 256, 0, stream>>>(fmap, Wd, bd, fd);
  pam_attn_kernel<<<(kB * kNT) / 8, 256, 0, stream>>>(fbT8, fcT8, (const uint32_t*)fd,
                                                      fmap, alpha, out);
}
